// GAMPNN_17763984736415
// MI455X (gfx1250) — matrix operand order repaired and checked by reading, not compiled, not run on hardware
//
#include <hip/hip_runtime.h>
#include <hip/hip_bf16.h>

#define NNODES 50000
#define NEDGES 400000
#define HIDD   128

// d_out layout (floats): h_new[50000*128] | x[50000*12] | ms0[400000] | ms1[400000] | att0[400000] | att1[400000]
#define OUT_X   6400000
#define OUT_MS  7000000
#define OUT_ATT 7800000

typedef _Float16 h16;
typedef __attribute__((ext_vector_type(16))) _Float16 v16h;
typedef __attribute__((ext_vector_type(8)))  float    v8f;

// ---------- parameter block passed by value to kernels ----------
struct GP {
  const float *h, *coord, *ea0, *ea1;
  const int   *ei0, *ei1;
  const float *rp_b1, *rp_b2, *pj_b1, *pj_b2;
  const float *ni_b, *mm_b1, *mm_b2, *em_b1, *em_b2, *nm_b1, *nm_b2, *cm_b1, *q_b, *k_b;
  const float *eps, *sig, *asig;
  float *out;
  // f32 workspace
  float *rsq, *esum, *esq, *denom, *cnt, *ssum, *aggp;
  unsigned *umax;
  float *rnorm, *emean, *einv;
  float *cdw, *radw, *biasw, *enw, *scorew, *exw, *alphaw;
  // f16 workspace
  h16 *efw, *hnew;
  const h16 *qW, *kW, *niW, *mm1W, *mm2W, *rp1W, *rp2W, *pj1W, *pj2W;
  const h16 *nm1W, *nm2W, *em1W, *em2W, *geoW, *cm1W, *cm2W;
};

__device__ __forceinline__ void lds_fence() {
  asm volatile("s_wait_dscnt 0" ::: "memory");
}
__device__ __forceinline__ float silu(float x) { return x / (1.f + __expf(-x)); }
__device__ __forceinline__ unsigned fenc(float f) {
  unsigned u = __float_as_uint(f);
  return (u & 0x80000000u) ? ~u : (u | 0x80000000u);
}
__device__ __forceinline__ float fdec(unsigned u) {
  return (u & 0x80000000u) ? __uint_as_float(u ^ 0x80000000u) : __uint_as_float(~u);
}

// ---------- WMMA tile helper: C[16 rows x 16 cols] += A(16xK) * W^T(Kx16) ----------
// A: row-major, stride lda (halfs).  W: row-major (out,in) padded, stride ldw; column n of B = row (ncol+n) of W.
__device__ __forceinline__ v8f wmma_acc(const h16* A, int lda, const h16* W, int ldw,
                                        int ncol, int ktiles, v8f c) {
  const int lane = threadIdx.x & 31;
  const int m = lane & 15;
  const int hi = (lane & 16) ? 1 : 0;
  for (int kt = 0; kt < ktiles; ++kt) {
    const int kc = kt * 32;
    v16h a, b;
    const h16* pa0 = A + m * lda + kc + hi * 8;
    const h16* pa1 = pa0 + 16;
#pragma unroll
    for (int i = 0; i < 8; ++i) { a[i] = pa0[i]; a[8 + i] = pa1[i]; }
    const h16* pb = W + (size_t)(ncol + m) * ldw + kc + hi * 8;
#pragma unroll
    for (int i = 0; i < 8; ++i) { b[i] = pb[i]; b[8 + i] = pb[16 + i]; }
    c = __builtin_amdgcn_wmma_f32_16x16x32_f16(false, a, false, b, (short)0, c, false, false);
  }
  return c;
}

// ---------- weight prep: f32 (rows x cols) -> f16 padded (rpad x kpad), zero-filled ----------
__global__ __launch_bounds__(256) void k_prep(const float* src, h16* dst,
                                              int rows, int cols, int kpad, int rpad) {
  int i = blockIdx.x * 256 + threadIdx.x;
  int tot = rpad * kpad;
  if (i >= tot) return;
  int r = i / kpad, c = i % kpad;
  dst[i] = (r < rows && c < cols) ? (h16)src[r * cols + c] : (h16)0.f;
}

// ---------- kernel 1: geometry, radial, LJ energy, gaussian bias, global stats ----------
__global__ __launch_bounds__(256) void k_geom(GP p) {
  const int type = blockIdx.y;
  const int i = blockIdx.x * 256 + threadIdx.x;
  const int* ei = type ? p.ei1 : p.ei0;
  __shared__ float sred[18];
  if (threadIdx.x < 18) sred[threadIdx.x] = 0.f;
  __syncthreads();
  if (i < NEDGES) {
    int row = ei[i], col = ei[NEDGES + i];
    float cd[12];
#pragma unroll
    for (int k = 0; k < 12; ++k) cd[k] = p.coord[(size_t)row * 12 + k] - p.coord[(size_t)col * 12 + k];
    size_t eb = (size_t)type * NEDGES + i;
#pragma unroll
    for (int k = 0; k < 12; ++k) p.cdw[eb * 12 + k] = cd[k];
#pragma unroll
    for (int a = 0; a < 4; ++a)
#pragma unroll
      for (int b = 0; b < 4; ++b) {
        float r = cd[a * 3] * cd[b * 3] + cd[a * 3 + 1] * cd[b * 3 + 1] + cd[a * 3 + 2] * cd[b * 3 + 2];
        p.radw[eb * 16 + a * 4 + b] = r;
        atomicAdd(&sred[a * 4 + b], r * r);
      }
    float d2 = cd[3] * cd[3] + cd[4] * cd[4] + cd[5] * cd[5];
    float dist = sqrtf(d2);
    float as = p.asig[0];
    p.biasw[eb] = __expf(-d2 / (2.f * as * as));
    float r = fmaxf(dist, 0.1f) + 1e-6f;
    float sr = p.sig[0] / r;
    float sr2 = sr * sr, sr6 = sr2 * sr2 * sr2;
    float en = 4.f * p.eps[0] * (sr6 * sr6 - sr6);
    p.enw[eb] = en;
    atomicAdd(&sred[16], en);
    atomicAdd(&sred[17], en * en);
  }
  __syncthreads();
  if (threadIdx.x < 16) atomicAdd(&p.rsq[type * 16 + threadIdx.x], sred[threadIdx.x]);
  else if (threadIdx.x == 16) atomicAdd(&p.esum[type], sred[16]);
  else if (threadIdx.x == 17) atomicAdd(&p.esq[type], sred[17]);
}

// ---------- kernel 2: finalize radial norm + energy mean/std ----------
__global__ __launch_bounds__(64) void k_fin(GP p) {
  int t = threadIdx.x;
  if (t < 32) p.rnorm[t] = 1.f / fmaxf(sqrtf(p.rsq[t]), 1e-12f);
  else if (t < 34) {
    int ty = t - 32;
    float mean = p.esum[ty] / (float)NEDGES;
    float var = (p.esq[ty] - (float)NEDGES * mean * mean) / ((float)NEDGES - 1.f);
    p.emean[ty] = mean;
    p.einv[ty] = 1.f / (sqrtf(fmaxf(var, 0.f)) + 1e-6f);
  }
}

// ---------- kernel 3: fused edge WMMA pipeline (score, ni, pos-enc, mm -> edge_feats) ----------
#define K3NW 2
__global__ __launch_bounds__(64) void k3_edge(GP p) {
  struct W {
    h16 xa[16 * 264];   // srctgt / rp scratch / enc / mm hidden
    h16 xb[16 * 200];   // tf / mf-input
    h16 pj[16 * 32];
    float cd3[16 * 3];
    int rowv[16], colv[16];
    float eav[16];
  };
  __shared__ W sw[K3NW];
  const int wid = threadIdx.x >> 5, lane = threadIdx.x & 31;
  W& w = sw[wid];
  const int type = blockIdx.y;
  const int e0 = (blockIdx.x * K3NW + wid) * 16;
  const int* ei = type ? p.ei1 : p.ei0;
  const float* ea = type ? p.ea1 : p.ea0;
  const int m = lane & 15, mb = (lane & 16) ? 8 : 0;
  if (lane == 0) { __builtin_prefetch(p.qW, 0, 2); __builtin_prefetch(p.niW, 0, 2); }
  if (lane < 16) {
    w.rowv[lane] = ei[e0 + lane];
    w.colv[lane] = ei[NEDGES + e0 + lane];
    w.eav[lane] = ea[e0 + lane];
  }
  lds_fence();
  const size_t eb = (size_t)type * NEDGES + e0;
  // stage [src | tgt] into xa
  for (int idx = lane; idx < 16 * 128; idx += 32) {
    int e = idx >> 7, f = idx & 127;
    w.xa[e * 264 + f]       = (h16)p.h[(size_t)w.rowv[e] * 128 + f];
    w.xa[e * 264 + 128 + f] = (h16)p.h[(size_t)w.colv[e] * 128 + f];
  }
  // stage tf = [rflat(16) | tgt(128) | ea | zeros] into xb
  for (int idx = lane; idx < 16 * 16; idx += 32) {
    int e = idx >> 4, j = idx & 15;
    w.xb[e * 200 + j] = (h16)(p.radw[(eb + e) * 16 + j] * p.rnorm[type * 16 + j]);
  }
  lds_fence();
  for (int idx = lane; idx < 16 * 128; idx += 32) {
    int e = idx >> 7, f = idx & 127;
    w.xb[e * 200 + 16 + f] = w.xa[e * 264 + 128 + f];
  }
  if (lane < 16) w.xb[lane * 200 + 144] = (h16)w.eav[lane];
  for (int idx = lane; idx < 16 * 55; idx += 32) {
    int e = idx / 55, j = idx % 55;
    w.xb[e * 200 + 145 + j] = (h16)0.f;
  }
  lds_fence();
  // ---- attention score = sum_n q*k + gauss bias; segment-max atomics ----
  float part[8];
#pragma unroll
  for (int r = 0; r < 8; ++r) part[r] = 0.f;
  for (int j = 0; j < 8; ++j) {
    v8f z0 = {}; v8f cq = wmma_acc(w.xa, 264, p.qW, 128, 16 * j, 4, z0);
    v8f z1 = {}; v8f ck = wmma_acc(w.xb, 200, p.kW, 160, 16 * j, 5, z1);
    float qb = p.q_b[16 * j + m], kb = p.k_b[16 * j + m];
#pragma unroll
    for (int r = 0; r < 8; ++r) part[r] += (cq[r] + qb) * silu(ck[r] + kb);
  }
#pragma unroll
  for (int r = 0; r < 8; ++r) {
    float v = part[r];
    v += __shfl_xor(v, 1, 32); v += __shfl_xor(v, 2, 32);
    v += __shfl_xor(v, 4, 32); v += __shfl_xor(v, 8, 32);
    part[r] = v;
  }
  if (m == 0) {
#pragma unroll
    for (int r = 0; r < 8; ++r) {
      int e = mb + r;
      float sc = part[r] + p.biasw[eb + e];
      p.scorew[eb + e] = sc;
      atomicMax(&p.umax[(size_t)type * NNODES + w.rowv[e]], fenc(sc));
    }
  }
  // ---- ni = silu([src,tgt] @ ni_w^T) -> xb cols 0..127 (mf input) ----
  for (int j = 0; j < 8; ++j) {
    v8f z = {}; v8f c = wmma_acc(w.xa, 264, p.niW, 256, 16 * j, 8, z);
    float b = p.ni_b[16 * j + m];
#pragma unroll
    for (int r = 0; r < 8; ++r) w.xb[(mb + r) * 200 + 16 * j + m] = (h16)silu(c[r] + b);
  }
  lds_fence();
  // ---- positional encoder: rp MLP -> cd3 -> enc -> pj MLP -> pe (xb cols 128..159) ----
  for (int idx = lane; idx < 16 * 32; idx += 32) {
    int e = idx >> 5, j = idx & 31;
    float v = (j < 16) ? p.radw[(eb + e) * 16 + j] * p.rnorm[type * 16 + j] : 0.f;
    w.xa[e * 264 + j] = (h16)v;
  }
  lds_fence();
  for (int j = 0; j < 8; ++j) {
    v8f z = {}; v8f c = wmma_acc(w.xa, 264, p.rp1W, 32, 16 * j, 1, z);
    float b = p.rp_b1[16 * j + m];
#pragma unroll
    for (int r = 0; r < 8; ++r) w.xa[(mb + r) * 264 + 32 + 16 * j + m] = (h16)silu(c[r] + b);
  }
  lds_fence();
  {
    v8f z = {}; v8f c = wmma_acc(&w.xa[32], 264, p.rp2W, 128, 0, 4, z);
    if (m < 3) {
      float b = p.rp_b2[m];
#pragma unroll
      for (int r = 0; r < 8; ++r) w.cd3[(mb + r) * 3 + m] = c[r] + b;
    }
  }
  lds_fence();
  for (int idx = lane; idx < 16 * 224; idx += 32) {
    int e = idx / 224, col = idx % 224;
    float c0 = w.cd3[e * 3], c1 = w.cd3[e * 3 + 1], c2 = w.cd3[e * 3 + 2];
    float d = sqrtf(c0 * c0 + c1 * c1 + c2 * c2);
    float val;
    if (col < 192) {
      int q = (col < 96) ? col : col - 96;
      int ch = q >> 5, f = q & 31;
      float cc = (ch == 0) ? c0 : ((ch == 1) ? c1 : c2);
      float fr = __expf((-1.f + 3.f * (float)f / 31.f) * 2.30258509299f);
      float sc = cc * fr;
      val = (col < 96) ? __sinf(sc) : __cosf(sc);
    } else if (col == 192) val = d;
    else if (col < 196) {
      int ch = col - 193;
      float cc = (ch == 0) ? c0 : ((ch == 1) ? c1 : c2);
      val = cc / (d + 1e-8f);
    } else val = 0.f;
    w.xa[e * 264 + col] = (h16)val;
  }
  lds_fence();
  for (int j = 0; j < 2; ++j) {
    v8f z = {}; v8f c = wmma_acc(w.xa, 264, p.pj1W, 224, 16 * j, 7, z);
    float b = p.pj_b1[16 * j + m];
#pragma unroll
    for (int r = 0; r < 8; ++r) w.pj[(mb + r) * 32 + 16 * j + m] = (h16)silu(c[r] + b);
  }
  lds_fence();
  for (int j = 0; j < 2; ++j) {
    v8f z = {}; v8f c = wmma_acc(w.pj, 32, p.pj2W, 32, 16 * j, 1, z);
    float b = p.pj_b2[16 * j + m];
#pragma unroll
    for (int r = 0; r < 8; ++r) w.xb[(mb + r) * 200 + 128 + 16 * j + m] = (h16)(c[r] + b);
  }
  if (lane < 16) w.xb[lane * 200 + 160] = (h16)w.eav[lane];
  lds_fence();
  // ---- mm MLP: mf = silu(silu([ni,pe,ea] @ W1) @ W2) -> edge_feats ----
  for (int j = 0; j < 8; ++j) {
    v8f z = {}; v8f c = wmma_acc(w.xb, 200, p.mm1W, 192, 16 * j, 6, z);
    float b = p.mm_b1[16 * j + m];
#pragma unroll
    for (int r = 0; r < 8; ++r) w.xa[(mb + r) * 264 + 16 * j + m] = (h16)silu(c[r] + b);
  }
  lds_fence();
  for (int j = 0; j < 8; ++j) {
    v8f z = {}; v8f c = wmma_acc(w.xa, 264, p.mm2W, 128, 16 * j, 4, z);
    float b = p.mm_b2[16 * j + m];
#pragma unroll
    for (int r = 0; r < 8; ++r) p.efw[(eb + mb + r) * 128 + 16 * j + m] = (h16)silu(c[r] + b);
  }
}

// ---------- kernel 4: softmax exp + denom ----------
__global__ __launch_bounds__(256) void k4_ex(GP p) {
  const int type = blockIdx.y;
  const int i = blockIdx.x * 256 + threadIdx.x;
  if (i >= NEDGES) return;
  const int* ei = type ? p.ei1 : p.ei0;
  int row = ei[i];
  size_t eb = (size_t)type * NEDGES + i;
  float mx = fdec(p.umax[(size_t)type * NNODES + row]);
  float ex = __expf(p.scorew[eb] - mx);
  p.exw[eb] = ex;
  atomicAdd(&p.denom[(size_t)type * NNODES + row], ex);
}

// ---------- kernel 5: alpha + att outputs ----------
__global__ __launch_bounds__(256) void k5_alpha(GP p) {
  const int type = blockIdx.y;
  const int i = blockIdx.x * 256 + threadIdx.x;
  if (i >= NEDGES) return;
  const int* ei = type ? p.ei1 : p.ei0;
  int row = ei[i];
  size_t eb = (size_t)type * NEDGES + i;
  float al = p.exw[eb] / p.denom[(size_t)type * NNODES + row];
  p.alphaw[eb] = al;
  p.out[OUT_ATT + (size_t)type * NEDGES + i] = al;
}

// ---------- kernel 6: coord model (cm WMMA) + scatter; node aggregation ----------
__global__ __launch_bounds__(128) void k6_coord(GP p) {
  struct W { h16 xb[16 * 136]; float w4[16 * 4]; int rowv[16]; };
  __shared__ W sw[4];
  const int wid = threadIdx.x >> 5, lane = threadIdx.x & 31;
  W& w = sw[wid];
  const int type = blockIdx.y;
  const int e0 = (blockIdx.x * 4 + wid) * 16;
  const int* ei = type ? p.ei1 : p.ei0;
  const int m = lane & 15, mb = (lane & 16) ? 8 : 0;
  if (lane < 16) w.rowv[lane] = ei[e0 + lane];
  lds_fence();
  const size_t eb = (size_t)type * NEDGES + e0;
  const h16* efA = p.efw + eb * 128;
  const h16* cm1 = p.cm1W + (size_t)type * 128 * 128;
  const h16* cm2 = p.cm2W + (size_t)type * 16 * 128;
  for (int j = 0; j < 8; ++j) {
    v8f z = {}; v8f c = wmma_acc(efA, 128, cm1, 128, 16 * j, 4, z);
    float b = p.cm_b1[type * 128 + 16 * j + m];
#pragma unroll
    for (int r = 0; r < 8; ++r) w.xb[(mb + r) * 136 + 16 * j + m] = (h16)silu(c[r] + b);
  }
  lds_fence();
  {
    v8f z = {}; v8f c = wmma_acc(w.xb, 136, cm2, 128, 0, 4, z);
    if (m < 4)
#pragma unroll
      for (int r = 0; r < 8; ++r) w.w4[(mb + r) * 4 + m] = c[r];
  }
  lds_fence();
  for (int idx = lane; idx < 192; idx += 32) {
    int e = idx / 12, k = idx % 12;
    float t = p.cdw[(eb + e) * 12 + k] * w.w4[e * 4 + (k / 3)];
    atomicAdd(&p.ssum[(size_t)w.rowv[e] * 12 + k], t);
  }
  if (lane < 16) atomicAdd(&p.cnt[w.rowv[lane]], 1.f);
  for (int idx = lane; idx < 2048; idx += 32) {
    int e = idx >> 7, f = idx & 127;
    float wg = type ? p.alphaw[eb + e] : 1.f;
    atomicAdd(&p.aggp[((size_t)type * NNODES + w.rowv[e]) * 128 + f], (float)efA[e * 128 + f] * wg);
  }
}

// ---------- kernel 7: node update (geo + nm WMMA) ----------
__global__ __launch_bounds__(128) void k7_node(GP p) {
  struct W { h16 xa[16 * 264]; h16 xb[16 * 136]; };
  __shared__ W sw[4];
  const int wid = threadIdx.x >> 5, lane = threadIdx.x & 31;
  W& w = sw[wid];
  const int tile = blockIdx.x * 4 + wid;
  if (tile * 16 >= NNODES) return;
  const int n0 = tile * 16;
  const int m = lane & 15, mb = (lane & 16) ? 8 : 0;
  for (int idx = lane; idx < 2048; idx += 32) {
    int e = idx >> 7, f = idx & 127;
    int node = n0 + e;
    w.xa[e * 264 + f]       = (h16)p.aggp[(size_t)node * 128 + f];
    w.xa[e * 264 + 128 + f] = (h16)p.aggp[((size_t)NNODES + node) * 128 + f];
  }
  lds_fence();
  for (int j = 0; j < 8; ++j) {
    v8f c = {};
    c = wmma_acc(w.xa, 264, p.geoW, 128, 16 * j, 4, c);
    c = wmma_acc(&w.xa[128], 264, p.geoW + 128 * 128, 128, 16 * j, 4, c);
#pragma unroll
    for (int r = 0; r < 8; ++r) w.xb[(mb + r) * 136 + 16 * j + m] = (h16)c[r];
  }
  lds_fence();
  for (int idx = lane; idx < 2048; idx += 32) {
    int e = idx >> 7, f = idx & 127;
    w.xa[e * 264 + f] = (h16)p.h[(size_t)(n0 + e) * 128 + f];
    w.xa[e * 264 + 128 + f] = w.xb[e * 136 + f];
  }
  lds_fence();
  for (int j = 0; j < 8; ++j) {
    v8f z = {}; v8f c = wmma_acc(w.xa, 264, p.nm1W, 256, 16 * j, 8, z);
    float b = p.nm_b1[16 * j + m];
#pragma unroll
    for (int r = 0; r < 8; ++r) w.xb[(mb + r) * 136 + 16 * j + m] = (h16)silu(c[r] + b);
  }
  lds_fence();
  for (int j = 0; j < 8; ++j) {
    v8f z = {}; v8f c = wmma_acc(w.xb, 136, p.nm2W, 128, 16 * j, 4, z);
    float b = p.nm_b2[16 * j + m];
#pragma unroll
    for (int r = 0; r < 8; ++r) {
      int node = n0 + mb + r, col = 16 * j + m;
      float hn = p.h[(size_t)node * 128 + col] + c[r] + b;
      p.out[(size_t)node * 128 + col] = hn;
      p.hnew[(size_t)node * 128 + col] = (h16)hn;
    }
  }
}

// ---------- kernel x: coordinate output ----------
__global__ __launch_bounds__(256) void kx_coord(GP p) {
  int i = blockIdx.x * 256 + threadIdx.x;
  if (i >= NNODES * 12) return;
  float c = p.cnt[i / 12];
  p.out[OUT_X + i] = p.coord[i] + p.ssum[i] / fmaxf(c, 1.f);
}

// ---------- kernel 8: edge model (em WMMA) ----------
__global__ __launch_bounds__(128) void k8_edge(GP p) {
  struct W { h16 xa[16 * 296]; h16 xb[16 * 136]; int rowv[16], colv[16]; };
  __shared__ W sw[4];
  const int wid = threadIdx.x >> 5, lane = threadIdx.x & 31;
  W& w = sw[wid];
  const int type = blockIdx.y;
  const int e0 = (blockIdx.x * 4 + wid) * 16;
  const int* ei = type ? p.ei1 : p.ei0;
  const float* ea = type ? p.ea1 : p.ea0;
  const int m = lane & 15, mb = (lane & 16) ? 8 : 0;
  if (lane < 16) { w.rowv[lane] = ei[e0 + lane]; w.colv[lane] = ei[NEDGES + e0 + lane]; }
  lds_fence();
  const size_t eb = (size_t)type * NEDGES + e0;
  for (int idx = lane; idx < 16 * 128; idx += 32) {
    int e = idx >> 7, f = idx & 127;
    w.xa[e * 296 + f]        = p.hnew[(size_t)w.rowv[e] * 128 + f];
    w.xa[e * 296 + 129 + f]  = p.hnew[(size_t)w.colv[e] * 128 + f];
  }
  if (lane < 16) {
    w.xa[lane * 296 + 128] = (h16)ea[e0 + lane];
    w.xa[lane * 296 + 257] = (h16)((p.enw[eb + lane] - p.emean[type]) * p.einv[type]);
  }
  for (int idx = lane; idx < 16 * 31; idx += 32) {
    int e = idx / 31, j = idx % 31;
    w.xa[e * 296 + 258 + j] = (h16)0.f;
  }
  lds_fence();
  for (int j = 0; j < 8; ++j) {
    v8f z = {}; v8f c = wmma_acc(w.xa, 296, p.em1W, 288, 16 * j, 9, z);
    float b = p.em_b1[16 * j + m];
#pragma unroll
    for (int r = 0; r < 8; ++r) w.xb[(mb + r) * 136 + 16 * j + m] = (h16)silu(c[r] + b);
  }
  lds_fence();
  {
    v8f z = {}; v8f c = wmma_acc(w.xb, 136, p.em2W, 128, 0, 4, z);
    if (m == 0) {
      float b = p.em_b2[0];
#pragma unroll
      for (int r = 0; r < 8; ++r)
        p.out[OUT_MS + (size_t)type * NEDGES + e0 + mb + r] = c[r] + b;
    }
  }
}

// =====================================================================
extern "C" void kernel_launch(void* const* d_in, const int* in_sizes, int n_in,
                              void* d_out, int out_size, void* d_ws, size_t ws_size,
                              hipStream_t stream) {
  (void)in_sizes; (void)n_in; (void)out_size;
  GP p;
  p.h     = (const float*)d_in[0];
  p.coord = (const float*)d_in[1];
  p.ea0   = (const float*)d_in[2];
  p.ea1   = (const float*)d_in[3];
  p.ei0   = (const int*)d_in[4];
  p.ei1   = (const int*)d_in[5];
  // params flattened in dict insertion order starting at index 8
  const float* rp_w1 = (const float*)d_in[8];   p.rp_b1 = (const float*)d_in[9];
  const float* rp_w2 = (const float*)d_in[10];  p.rp_b2 = (const float*)d_in[11];
  const float* pj_w1 = (const float*)d_in[12];  p.pj_b1 = (const float*)d_in[13];
  const float* pj_w2 = (const float*)d_in[14];  p.pj_b2 = (const float*)d_in[15];
  const float* ni_w  = (const float*)d_in[16];  p.ni_b  = (const float*)d_in[17];
  const float* mm_w1 = (const float*)d_in[18];  p.mm_b1 = (const float*)d_in[19];
  const float* mm_w2 = (const float*)d_in[20];  p.mm_b2 = (const float*)d_in[21];
  const float* em_w1 = (const float*)d_in[22];  p.em_b1 = (const float*)d_in[23];
  const float* em_w2 = (const float*)d_in[24];  p.em_b2 = (const float*)d_in[25];
  const float* nm_w1 = (const float*)d_in[26];  p.nm_b1 = (const float*)d_in[27];
  const float* nm_w2 = (const float*)d_in[28];  p.nm_b2 = (const float*)d_in[29];
  const float* geo_w = (const float*)d_in[30];
  const float* cm_w1 = (const float*)d_in[31];  p.cm_b1 = (const float*)d_in[32];
  const float* cm_w2 = (const float*)d_in[33];
  const float* q_w   = (const float*)d_in[34];  p.q_b   = (const float*)d_in[35];
  const float* k_w   = (const float*)d_in[36];  p.k_b   = (const float*)d_in[37];
  // v_w / v_b (38,39) are dead in the reference graph
  p.eps  = (const float*)d_in[40];
  p.sig  = (const float*)d_in[41];
  p.asig = (const float*)d_in[42];
  p.out  = (float*)d_out;

  // ---- workspace carve-up ----
  char* base = (char*)d_ws;
  size_t off = 0;
  auto alloc = [&](size_t bytes) -> char* {
    char* r = base + off;
    off += (bytes + 255) & ~(size_t)255;
    return r;
  };
  const size_t E = NEDGES, N = NNODES;
  // zero-init region (one memset)
  char* zbeg = base + off;
  p.rsq   = (float*)alloc(32 * 4);
  p.esum  = (float*)alloc(2 * 4);
  p.esq   = (float*)alloc(2 * 4);
  p.umax  = (unsigned*)alloc(2 * N * 4);
  p.denom = (float*)alloc(2 * N * 4);
  p.cnt   = (float*)alloc(N * 4);
  p.ssum  = (float*)alloc(N * 12 * 4);
  p.aggp  = (float*)alloc(2 * N * 128 * 4);
  size_t zbytes = (size_t)((base + off) - zbeg);
  // non-zeroed region
  p.rnorm  = (float*)alloc(32 * 4);
  p.emean  = (float*)alloc(2 * 4);
  p.einv   = (float*)alloc(2 * 4);
  p.cdw    = (float*)alloc(2 * E * 12 * 4);
  p.radw   = (float*)alloc(2 * E * 16 * 4);
  p.biasw  = (float*)alloc(2 * E * 4);
  p.enw    = (float*)alloc(2 * E * 4);
  p.scorew = (float*)alloc(2 * E * 4);
  p.exw    = (float*)alloc(2 * E * 4);
  p.alphaw = (float*)alloc(2 * E * 4);
  p.efw    = (h16*)alloc(2 * E * 128 * 2);
  p.hnew   = (h16*)alloc(N * 128 * 2);
  h16* qW   = (h16*)alloc(128 * 128 * 2);
  h16* kW   = (h16*)alloc(128 * 160 * 2);
  h16* niW  = (h16*)alloc(128 * 256 * 2);
  h16* mm1W = (h16*)alloc(128 * 192 * 2);
  h16* mm2W = (h16*)alloc(128 * 128 * 2);
  h16* rp1W = (h16*)alloc(128 * 32 * 2);
  h16* rp2W = (h16*)alloc(16 * 128 * 2);
  h16* pj1W = (h16*)alloc(32 * 224 * 2);
  h16* pj2W = (h16*)alloc(32 * 32 * 2);
  h16* nm1W = (h16*)alloc(128 * 256 * 2);
  h16* nm2W = (h16*)alloc(128 * 128 * 2);
  h16* em1W = (h16*)alloc(128 * 288 * 2);
  h16* em2W = (h16*)alloc(16 * 128 * 2);
  h16* geoW = (h16*)alloc(2 * 128 * 128 * 2);
  h16* cm1W = (h16*)alloc(2 * 128 * 128 * 2);
  h16* cm2W = (h16*)alloc(2 * 16 * 128 * 2);
  p.qW = qW; p.kW = kW; p.niW = niW; p.mm1W = mm1W; p.mm2W = mm2W;
  p.rp1W = rp1W; p.rp2W = rp2W; p.pj1W = pj1W; p.pj2W = pj2W;
  p.nm1W = nm1W; p.nm2W = nm2W; p.em1W = em1W; p.em2W = em2W;
  p.geoW = geoW; p.cm1W = cm1W; p.cm2W = cm2W;
  (void)ws_size;

  hipMemsetAsync(zbeg, 0, zbytes, stream);

  auto prep = [&](const float* src, h16* dst, int rows, int cols, int kpad, int rpad) {
    int tot = rpad * kpad;
    k_prep<<<(tot + 255) / 256, 256, 0, stream>>>(src, dst, rows, cols, kpad, rpad);
  };
  prep(q_w,   qW,   128, 128, 128, 128);
  prep(k_w,   kW,   128, 145, 160, 128);
  prep(ni_w,  niW,  128, 256, 256, 128);
  prep(mm_w1, mm1W, 128, 161, 192, 128);
  prep(mm_w2, mm2W, 128, 128, 128, 128);
  prep(rp_w1, rp1W, 128, 16,  32,  128);
  prep(rp_w2, rp2W, 3,   128, 128, 16);
  prep(pj_w1, pj1W, 32,  196, 224, 32);
  prep(pj_w2, pj2W, 32,  32,  32,  32);
  prep(nm_w1, nm1W, 128, 256, 256, 128);
  prep(nm_w2, nm2W, 128, 128, 128, 128);
  prep(em_w1, em1W, 128, 258, 288, 128);
  prep(em_w2, em2W, 1,   128, 128, 16);
  prep(geo_w, geoW, 256, 128, 128, 256);
  prep(cm_w1, cm1W, 256, 128, 128, 256);
  prep(cm_w2,             cm2W,            4, 128, 128, 16);
  prep(cm_w2 + 4 * 128,   cm2W + 16 * 128, 4, 128, 128, 16);

  dim3 gE((NEDGES + 255) / 256, 2);
  k_geom<<<gE, 256, 0, stream>>>(p);
  k_fin<<<1, 64, 0, stream>>>(p);
  k3_edge<<<dim3(NEDGES / (16 * K3NW), 2), 64, 0, stream>>>(p);
  k4_ex<<<gE, 256, 0, stream>>>(p);
  k5_alpha<<<gE, 256, 0, stream>>>(p);
  k6_coord<<<dim3(NEDGES / 64, 2), 128, 0, stream>>>(p);
  k7_node<<<(NNODES / 16 + 3) / 4, 128, 0, stream>>>(p);
  kx_coord<<<(NNODES * 12 + 255) / 256, 256, 0, stream>>>(p);
  k8_edge<<<dim3(NEDGES / 64, 2), 128, 0, stream>>>(p);
}
